// RelationAttention_19851338842357
// MI455X (gfx1250) — matrix operand order repaired and checked by reading, not compiled, not run on hardware
//
#include <hip/hip_runtime.h>
#include <hip/hip_bf16.h>

#define LQ 1024
#define BZ 8
#define DD 512
#define NH 8
#define DH 64

typedef __attribute__((ext_vector_type(16))) __bf16 v16bf;
typedef __attribute__((ext_vector_type(8)))  __bf16 v8bf;
typedef __attribute__((ext_vector_type(8)))  float  v8f;

__device__ __forceinline__ v16bf cat8(v8bf lo, v8bf hi) {
  return __builtin_shufflevector(lo, hi, 0,1,2,3,4,5,6,7,8,9,10,11,12,13,14,15);
}

// A-fragment (16x32 bf16, M x K): lane L<16 holds row L, K {0..7,16..23};
// lane 16+L holds row L, K {8..15,24..31}.  `p` must already be
// rowptr + k0 + (lane>=16 ? 8 : 0).
__device__ __forceinline__ v16bf load_a_frag(const __bf16* p) {
  v8bf lo = *reinterpret_cast<const v8bf*>(p);
  v8bf hi = *reinterpret_cast<const v8bf*>(p + 16);
  return cat8(lo, hi);
}

// B-fragment (32x16 bf16, K x N): lane N<16 holds column N, K {0..7,16..23};
// lane 16+N holds column N, K {8..15,24..31}.  `p` = colptr + k0 + (lane>=16 ? 16 : 0),
// where successive K values are contiguous in memory (B column == source row).
__device__ __forceinline__ v16bf load_b_frag(const __bf16* p) {
  const __bf16* q = p - (((threadIdx.x & 31) >> 4) ? 8 : 0);
  v8bf lo = *reinterpret_cast<const v8bf*>(q);
  v8bf hi = *reinterpret_cast<const v8bf*>(q + 16);
  return cat8(lo, hi);
}

__device__ __forceinline__ v8f wmma_bf16(v16bf a, v16bf b, v8f c) {
  return __builtin_amdgcn_wmma_f32_16x16x32_bf16(false, a, false, b,
                                                 (short)0, c, false, false);
}

// ---------------------------------------------------------------------------
// Kernel 1: h = transpose(query) -> bf16; hn = normalized rows -> bf16
// one warp per (b, l) row of 512
// ---------------------------------------------------------------------------
__global__ void prep_h_kernel(const float* __restrict__ query,
                              __bf16* __restrict__ hbf,
                              __bf16* __restrict__ hnbf) {
  int warp = (blockIdx.x * blockDim.x + threadIdx.x) >> 5;
  int lane = threadIdx.x & 31;
  if (warp >= BZ * LQ) return;
  int b = warp >> 10;
  int l = warp & (LQ - 1);
  const float* row = query + ((size_t)l * BZ + b) * DD;
  float v[16];
  float ss = 0.f;
#pragma unroll
  for (int t = 0; t < 16; ++t) {
    v[t] = row[lane + 32 * t];
    ss += v[t] * v[t];
  }
#pragma unroll
  for (int off = 16; off >= 1; off >>= 1) ss += __shfl_xor(ss, off, 32);
  float inv = 1.0f / fmaxf(sqrtf(ss), 1e-8f);
  __bf16* hb = hbf  + ((size_t)b * LQ + l) * DD;
  __bf16* hn = hnbf + ((size_t)b * LQ + l) * DD;
#pragma unroll
  for (int t = 0; t < 16; ++t) {
    int k = lane + 32 * t;
    hb[k] = (__bf16)v[t];
    hn[k] = (__bf16)(v[t] * inv);
  }
}

// ---------------------------------------------------------------------------
// Kernel 2: weights fp32 -> bf16 (row-major, consumed as WMMA B columns)
// ---------------------------------------------------------------------------
__global__ void prep_w_kernel(const float* __restrict__ Wq,
                              const float* __restrict__ Wk,
                              const float* __restrict__ Wv,
                              __bf16* __restrict__ wbf) {
  int i = blockIdx.x * blockDim.x + threadIdx.x;
  const int n = DD * DD;
  if (i < n) {
    wbf[i]         = (__bf16)Wq[i];
    wbf[n + i]     = (__bf16)Wk[i];
    wbf[2 * n + i] = (__bf16)Wv[i];
  }
}

// ---------------------------------------------------------------------------
// Kernel 3: Q/K/V = h @ W^T + bias.  One warp per 16x64 output strip
// (4 accumulators, A-fragment reused 4x => ~3x less fragment traffic).
// A 64-wide strip == one head, so `head` is warp-uniform.
// Q,K stored [b][h][Lq][64] bf16; V stored transposed [b][h][64][Lq] bf16.
// ---------------------------------------------------------------------------
__global__ void qkv_gemm_kernel(const __bf16* __restrict__ hbf,
                                const __bf16* __restrict__ wbf,
                                const float* __restrict__ bq,
                                const float* __restrict__ bk,
                                const float* __restrict__ bv,
                                __bf16* __restrict__ Qb,
                                __bf16* __restrict__ Kb,
                                __bf16* __restrict__ Vtb) {
  int warp = (blockIdx.x * blockDim.x + threadIdx.x) >> 5;
  int lane = threadIdx.x & 31;
  int nt = warp & 7;   warp >>= 3;   // 8 strips of 64 cols (== head index)
  int mt = warp & 63;  warp >>= 6;   // 64 row tiles
  int b  = warp & 7;   warp >>= 3;   // 8 batches
  int mat = warp;                    // 0:Q 1:K 2:V
  if (mat >= 3) return;
  int m0 = mt * 16, n0 = nt * 64;
  int lhalf = lane >> 4, lmod = lane & 15;

  const __bf16* Arow = hbf + ((size_t)b * LQ + m0 + lmod) * DD + (lhalf ? 8 : 0);
  const __bf16* Bcol = wbf + (size_t)mat * DD * DD
                           + (size_t)(n0 + lmod) * DD + (lhalf ? 16 : 0);
  v8f acc[4] = {};
#pragma unroll 2
  for (int k0 = 0; k0 < DD; k0 += 32) {
    v16bf af = load_a_frag(Arow + k0);
#pragma unroll
    for (int sub = 0; sub < 4; ++sub) {
      v16bf bf = load_b_frag(Bcol + (size_t)sub * 16 * DD + k0);
      acc[sub] = wmma_bf16(af, bf, acc[sub]);
    }
  }
  const float* bias = (mat == 0) ? bq : (mat == 1) ? bk : bv;
  int head = nt;  // n0 is 64-aligned: whole strip lives in one head
#pragma unroll
  for (int sub = 0; sub < 4; ++sub) {
    int dc = sub * 16 + lmod;
    float bv_ = bias[n0 + dc];
#pragma unroll
    for (int r = 0; r < 8; ++r) {
      int row = m0 + r + (lhalf ? 8 : 0);
      float v = acc[sub][r] + bv_;
      if (mat == 0)
        Qb[(((size_t)b * NH + head) * LQ + row) * DH + dc] = (__bf16)v;
      else if (mat == 1)
        Kb[(((size_t)b * NH + head) * LQ + row) * DH + dc] = (__bf16)v;
      else
        Vtb[(((size_t)b * NH + head) * DH + dc) * LQ + row] = (__bf16)v;
    }
  }
}

// ---------------------------------------------------------------------------
// Kernel 4: mask[b][i][j] = !adj.  cos via WMMA over hn, IoU from segments.
// One warp per 16x64 tile of the 1024x1024 mask (4 accs, A reused 4x).
// ---------------------------------------------------------------------------
__global__ void build_mask_kernel(const __bf16* __restrict__ hnbf,
                                  const float* __restrict__ segments,
                                  unsigned char* __restrict__ mask) {
  int warp = (blockIdx.x * blockDim.x + threadIdx.x) >> 5;
  int lane = threadIdx.x & 31;
  int jt = warp & 15;  warp >>= 4;   // 16 strips of 64 cols
  int it = warp & 63;  warp >>= 6;   // 64 row tiles
  int b  = warp;
  if (b >= BZ) return;
  int i0 = it * 16, j0 = jt * 64;
  int lhalf = lane >> 4, lmod = lane & 15;

  const __bf16* Arow = hnbf + ((size_t)b * LQ + i0 + lmod) * DD + (lhalf ? 8 : 0);
  const __bf16* Bcol = hnbf + ((size_t)b * LQ + j0 + lmod) * DD + (lhalf ? 16 : 0);
  v8f acc[4] = {};
#pragma unroll 2
  for (int k0 = 0; k0 < DD; k0 += 32) {
    v16bf af = load_a_frag(Arow + k0);
#pragma unroll
    for (int sub = 0; sub < 4; ++sub) {
      v16bf bf = load_b_frag(Bcol + (size_t)sub * 16 * DD + k0);
      acc[sub] = wmma_bf16(af, bf, acc[sub]);   // cos[i0+r(+8)][j0+sub*16+lmod]
    }
  }
  const float* seg = segments + (size_t)b * LQ * 2;
#pragma unroll
  for (int sub = 0; sub < 4; ++sub) {
    int j = j0 + sub * 16 + lmod;
    float cj = seg[2 * j], lj = seg[2 * j + 1];
    float sj = cj - 0.5f * lj, ej = cj + 0.5f * lj;
#pragma unroll
    for (int r = 0; r < 8; ++r) {
      int i = i0 + r + (lhalf ? 8 : 0);
      float ci = seg[2 * i], li = seg[2 * i + 1];
      float si = ci - 0.5f * li, ei = ci + 0.5f * li;
      float inter = fmaxf(fminf(ei, ej) - fmaxf(si, sj), 0.f);
      float iou = inter / (li + lj - inter);
      bool adj = (i == j);
      if (iou <= 0.2f) adj = true;
      if (acc[sub][r] <= 0.2f) adj = false;
      mask[((size_t)b * LQ + i) * LQ + j] = adj ? 0 : 1;  // 1 == masked (-inf)
    }
  }
}

// ---------------------------------------------------------------------------
// Kernel 5: flash attention.  One warp per (b, head, 16-query tile),
// streaming 32-key chunks; online softmax; P repacked via per-warp LDS tile.
// Writes final output (attn@V + h) directly in [Lq, bz, d] fp32.
// ---------------------------------------------------------------------------
__global__ void flash_attn_kernel(const __bf16* __restrict__ Qb,
                                  const __bf16* __restrict__ Kb,
                                  const __bf16* __restrict__ Vtb,
                                  const unsigned char* __restrict__ mask,
                                  const float* __restrict__ query,
                                  float* __restrict__ out) {
  __shared__ __align__(32) __bf16 ptile[8][16 * 32];  // per-warp P tile
  int wid  = threadIdx.x >> 5;
  int lane = threadIdx.x & 31;
  int warp = blockIdx.x * (blockDim.x >> 5) + wid;
  int qt = warp & 63;  warp >>= 6;
  int h  = warp & 7;   warp >>= 3;
  int b  = warp;
  if (b >= BZ) return;
  int q0 = qt * 16;
  int lhalf = lane >> 4, lmod = lane & 15;

  const __bf16* Qrow  = Qb  + (((size_t)b * NH + h) * LQ + q0 + lmod) * DH + (lhalf ? 8 : 0);
  const __bf16* Kbase = Kb  + (((size_t)b * NH + h) * LQ) * DH;
  const __bf16* Vbase = Vtb + (((size_t)b * NH + h) * DH) * LQ;
  const unsigned char* Mbase = mask + (size_t)b * LQ * LQ;

  v16bf aq[2];
#pragma unroll
  for (int c = 0; c < 2; ++c) aq[c] = load_a_frag(Qrow + 32 * c);

  float m_i[8], l_i[8];
  v8f oacc[4] = {};
#pragma unroll
  for (int r = 0; r < 8; ++r) { m_i[r] = -1e30f; l_i[r] = 0.f; }
  __bf16* myp = &ptile[wid][0];

  for (int j0 = 0; j0 < LQ; j0 += 32) {
    if (j0 + 32 < LQ) {  // stream next K/V chunks toward the caches
      __builtin_prefetch(Kbase + (size_t)(j0 + 32 + lmod) * DH, 0, 1);
      __builtin_prefetch(Vbase + (size_t)lmod * LQ + j0 + 32, 0, 1);
    }

    // S = Q K^T for 32 keys (two 16-key N-subtiles, two dh-32 K-chunks)
    v8f s[2]; s[0] = v8f{}; s[1] = v8f{};
#pragma unroll
    for (int sub = 0; sub < 2; ++sub)
#pragma unroll
      for (int c = 0; c < 2; ++c) {
        const __bf16* brow = Kbase + (size_t)(j0 + sub * 16 + lmod) * DH
                                   + c * 32 + (lhalf ? 16 : 0);
        s[sub] = wmma_bf16(aq[c], load_b_frag(brow), s[sub]);
      }

    // mask + online softmax (rows live in 16-lane halves; xor 1/2/4/8 reduce)
    float p[2][8], mnew[8];
#pragma unroll
    for (int r = 0; r < 8; ++r) {
      int i = q0 + r + (lhalf ? 8 : 0);
      float v0 = s[0][r] * 0.125f;   // 1/sqrt(64)
      float v1 = s[1][r] * 0.125f;
      if (Mbase[(size_t)i * LQ + j0 + lmod])      v0 = -1e30f;
      if (Mbase[(size_t)i * LQ + j0 + 16 + lmod]) v1 = -1e30f;
      p[0][r] = v0; p[1][r] = v1;
      float mx = fmaxf(v0, v1);
#pragma unroll
      for (int off = 1; off < 16; off <<= 1) mx = fmaxf(mx, __shfl_xor(mx, off, 32));
      mnew[r] = fmaxf(m_i[r], mx);
    }
#pragma unroll
    for (int r = 0; r < 8; ++r) {
      float alpha = __expf(m_i[r] - mnew[r]);
      float p0 = (p[0][r] <= -1e29f) ? 0.f : __expf(p[0][r] - mnew[r]);
      float p1 = (p[1][r] <= -1e29f) ? 0.f : __expf(p[1][r] - mnew[r]);
      float rs = p0 + p1;
#pragma unroll
      for (int off = 1; off < 16; off <<= 1) rs += __shfl_xor(rs, off, 32);
      l_i[r] = l_i[r] * alpha + rs;
      m_i[r] = mnew[r];
      p[0][r] = p0; p[1][r] = p1;
#pragma unroll
      for (int n = 0; n < 4; ++n) oacc[n][r] *= alpha;
    }

    // P (C/D layout) -> LDS tile [row][key] -> A-fragment
#pragma unroll
    for (int sub = 0; sub < 2; ++sub)
#pragma unroll
      for (int r = 0; r < 8; ++r) {
        int row = r + (lhalf ? 8 : 0);
        myp[row * 32 + sub * 16 + lmod] = (__bf16)p[sub][r];
      }
    asm volatile("s_wait_dscnt 0" ::: "memory");  // cross-lane LDS RAW fence
    v16bf pf = load_a_frag(myp + lmod * 32 + (lhalf ? 8 : 0));

    // out += P @ V  (V stored transposed => contiguous B columns)
#pragma unroll
    for (int n = 0; n < 4; ++n) {
      const __bf16* brow = Vbase + (size_t)(n * 16 + lmod) * LQ + j0 + (lhalf ? 16 : 0);
      oacc[n] = wmma_bf16(pf, load_b_frag(brow), oacc[n]);
    }
  }

  // epilogue: normalize, add residual h (== query transposed), write output
#pragma unroll
  for (int r = 0; r < 8; ++r) {
    int row = q0 + r + (lhalf ? 8 : 0);
    float invl = 1.f / l_i[r];
    const float* qrow = query + ((size_t)row * BZ + b) * DD;
    float* orow = out + ((size_t)row * BZ + b) * DD;
#pragma unroll
    for (int n = 0; n < 4; ++n) {
      int dcol = h * DH + n * 16 + lmod;
      orow[dcol] = oacc[n][r] * invl + qrow[dcol];
    }
  }
}

// ---------------------------------------------------------------------------
extern "C" void kernel_launch(void* const* d_in, const int* in_sizes, int n_in,
                              void* d_out, int out_size, void* d_ws, size_t ws_size,
                              hipStream_t stream) {
  const float* query    = (const float*)d_in[0];   // [Lq, bz, d]
  const float* segments = (const float*)d_in[1];   // [bz, Lq, 2]
  const float* Wq = (const float*)d_in[2];
  const float* bq = (const float*)d_in[3];
  const float* Wk = (const float*)d_in[4];
  const float* bk = (const float*)d_in[5];
  const float* Wv = (const float*)d_in[6];
  const float* bv = (const float*)d_in[7];
  float* out = (float*)d_out;

  char* ws = (char*)d_ws;
  size_t off = 0;
  const size_t HB = (size_t)BZ * LQ * DD * 2;      // 8 MB bf16 arrays
  __bf16* hbf  = (__bf16*)(ws + off); off += HB;
  __bf16* hnbf = (__bf16*)(ws + off); off += HB;
  __bf16* wbf  = (__bf16*)(ws + off); off += (size_t)3 * DD * DD * 2;
  __bf16* Qb   = (__bf16*)(ws + off); off += HB;
  __bf16* Kb   = (__bf16*)(ws + off); off += HB;
  __bf16* Vtb  = (__bf16*)(ws + off); off += HB;
  unsigned char* mask = (unsigned char*)(ws + off); off += (size_t)BZ * LQ * LQ;

  // 1) transpose + normalize (8192 warps)
  prep_h_kernel<<<1024, 256, 0, stream>>>(query, hbf, hnbf);
  // 2) weights to bf16
  prep_w_kernel<<<(DD * DD + 255) / 256, 256, 0, stream>>>(Wq, Wk, Wv, wbf);
  // 3) QKV GEMM: 3*8*64*8 = 12288 warps, 8 warps/block
  qkv_gemm_kernel<<<1536, 256, 0, stream>>>(hbf, wbf, bq, bk, bv, Qb, Kb, Vtb);
  // 4) mask: 8*64*16 = 8192 warps
  build_mask_kernel<<<1024, 256, 0, stream>>>(hnbf, segments, mask);
  // 5) flash attention: 8*8*64 = 4096 warps
  flash_attn_kernel<<<512, 256, 0, stream>>>(Qb, Kb, Vtb, mask, query, out);
}
